// EnhancedMoETransformer_87376814670607
// MI455X (gfx1250) — compile-verified, matrix operand order checked by reading, not run on hardware
//
#include <hip/hip_runtime.h>
#include <hip/hip_bf16.h>
#include <math.h>

// ---------------------------------------------------------------------------
// EnhancedMoETransformer forward for MI455X (gfx1250, wave32, WMMA).
// GEMMs: v_wmma_f32_16x16x32_bf16, fp32 accum. Tiles staged into LDS with
// GLOBAL_LOAD_ASYNC_TO_LDS_B128 (double buffered, ASYNCcnt pipelined) when the
// toolchain exposes the builtins; synchronous fallback otherwise.
// ---------------------------------------------------------------------------

typedef __bf16 bf16;
typedef __attribute__((ext_vector_type(16))) __bf16 v16bf;
typedef __attribute__((ext_vector_type(8)))  float  v8f;
typedef __attribute__((__vector_size__(16))) int    v4i_;

#ifndef __has_builtin
#define __has_builtin(x) 0
#endif
#if defined(__AMDGCN__) &&                                        \
    __has_builtin(__builtin_amdgcn_global_load_async_to_lds_b128) && \
    __has_builtin(__builtin_amdgcn_s_wait_asynccnt)
#define ASYNC_OK 1
#else
#define ASYNC_OK 0
#endif

#define BM 128
#define BN 64
#define BKK 32
#define LDP (BKK + 8)  // 160B row stride: 16B-aligned, staggers LDS banks

__device__ __forceinline__ float gelu_exact(float x) {
  return 0.5f * x * (1.0f + erff(x * 0.70710678118654752f));
}

template <int N>
__device__ __forceinline__ void wait_async() {
#if ASYNC_OK
  __builtin_amdgcn_s_wait_asynccnt(N);
#endif
}

__device__ __forceinline__ void async_b128(const float* g, const float* l) {
#if ASYNC_OK
  __builtin_amdgcn_global_load_async_to_lds_b128(
      (__attribute__((address_space(1))) v4i_*)(unsigned long long)(uintptr_t)g,
      (__attribute__((address_space(3))) v4i_*)(unsigned)(uintptr_t)l, 0, 0);
#else
  (void)g; (void)l;
#endif
}

// Build a 16-wide bf16 WMMA fragment from an fp32 LDS row.
// ISA 16-bit A/B layout: lane half h reads K in [h*8, h*8+7] then
// [16+h*8, 16+h*8+7] -> four b128 DS loads + packed converts.
__device__ __forceinline__ v16bf cvt_frag(const float* p, int half) {
  const float4 a0 = *(const float4*)(p + half * 8);
  const float4 a1 = *(const float4*)(p + half * 8 + 4);
  const float4 b0 = *(const float4*)(p + 16 + half * 8);
  const float4 b1 = *(const float4*)(p + 20 + half * 8);
  v16bf r;
  r[0] = (bf16)a0.x;  r[1] = (bf16)a0.y;  r[2] = (bf16)a0.z;  r[3] = (bf16)a0.w;
  r[4] = (bf16)a1.x;  r[5] = (bf16)a1.y;  r[6] = (bf16)a1.z;  r[7] = (bf16)a1.w;
  r[8] = (bf16)b0.x;  r[9] = (bf16)b0.y;  r[10] = (bf16)b0.z; r[11] = (bf16)b0.w;
  r[12] = (bf16)b1.x; r[13] = (bf16)b1.y; r[14] = (bf16)b1.z; r[15] = (bf16)b1.w;
  return r;
}

// C[m,n] = beta*C + rowscale[m] * act(alpha * sum_k A[m,k]*Wmat(n,k) + bias[n])
// Wmat(n,k) = WT ? W[k*ldw + n] : W[n*ldw + k]
// Batched over blockIdx.z with two-level (b,h) strides; H2 >= 1.
template <bool WT, bool GELU>
__global__ __launch_bounds__(128) void gemm_bf16_wmma(
    const float* __restrict__ A, const float* __restrict__ W,
    const float* __restrict__ bias,
    const float* __restrict__ rowscale, int rsStride,
    float* __restrict__ C,
    int M, int N, int K, int lda, int ldw, int ldc,
    float alpha, float beta, int H2,
    long long sAb, long long sAh, long long sWb, long long sWh,
    long long sCb, long long sCh) {
  __shared__ __attribute__((aligned(16))) float Asf[2][BM][LDP];
  __shared__ __attribute__((aligned(16))) float Wsf[2][BN][LDP];

  const int tid  = threadIdx.x;
  const int lane = tid & 31;
  const int wave = tid >> 5;
  const int half = lane >> 4;
  const int l16  = lane & 15;

  const int z  = blockIdx.z;
  const int zb = z / H2, zh = z % H2;
  A += zb * sAb + zh * sAh;
  W += zb * sWb + zh * sWh;
  C += zb * sCb + zh * sCh;

  const int m0 = blockIdx.y * BM;
  const int n0 = blockIdx.x * BN;

  v8f acc[8];
#pragma unroll
  for (int i = 0; i < 8; ++i) acc[i] = v8f{0, 0, 0, 0, 0, 0, 0, 0};

  // ---- tile staging (fp32 into LDS) ----
  auto stage = [&](int sb, int kk) {
#if ASYNC_OK
    // A tile: each wave DMAs its own 32-row slab, 16B per lane per op.
#pragma unroll
    for (int i = 0; i < 8; ++i) {
      const int r = wave * 32 + i * 4 + (lane >> 3);
      const int c = (lane & 7) * 4;
      async_b128(A + (long long)(m0 + r) * lda + kk + c, &Asf[sb][r][c]);
    }
    if (!WT) {
#pragma unroll
      for (int i = 0; i < 4; ++i) {
        const int r = wave * 16 + i * 4 + (lane >> 3);
        const int c = (lane & 7) * 4;
        async_b128(W + (long long)(n0 + r) * ldw + kk + c, &Wsf[sb][r][c]);
      }
    }
#else
    {
      const int c = (tid & 7) * 4;
      for (int r = tid >> 3; r < BM; r += 16) {
        *(float4*)&Asf[sb][r][c] =
            *(const float4*)(A + (long long)(m0 + r) * lda + kk + c);
      }
    }
    if (!WT) {
      const int c = (tid & 7) * 4;
      for (int r = tid >> 3; r < BN; r += 16) {
        *(float4*)&Wsf[sb][r][c] =
            *(const float4*)(W + (long long)(n0 + r) * ldw + kk + c);
      }
    }
#endif
    if (WT) {  // transposed B (P@V): must swizzle, stays on the sync path
      const int nn = (tid & 15) * 4;
      for (int kr = tid >> 4; kr < BKK; kr += 8) {
        const float4 f = *(const float4*)(W + (long long)(kk + kr) * ldw + n0 + nn);
        Wsf[sb][nn + 0][kr] = f.x;
        Wsf[sb][nn + 1][kr] = f.y;
        Wsf[sb][nn + 2][kr] = f.z;
        Wsf[sb][nn + 3][kr] = f.w;
      }
      __builtin_prefetch(W + (long long)(kk + BKK + (tid >> 4)) * ldw + n0, 0, 1);
    }
  };

  // ---- double-buffered ASYNCcnt pipeline ----
  int buf = 0;
  stage(0, 0);
  for (int kk = 0; kk < K; kk += BKK) {
    const bool hasNext = (kk + BKK) < K;
    if (hasNext) stage(buf ^ 1, kk + BKK);
    // oldest (current-tile) DMAs must have landed; next tile may stay in flight
    if (hasNext) wait_async<WT ? 8 : 12>();
    else         wait_async<0>();
    __syncthreads();

    const v16bf a0 = cvt_frag(&Asf[buf][wave * 32 + l16][0], half);
    const v16bf a1 = cvt_frag(&Asf[buf][wave * 32 + 16 + l16][0], half);
#pragma unroll
    for (int j = 0; j < 4; ++j) {
      const v16bf b = cvt_frag(&Wsf[buf][j * 16 + l16][0], half);
      acc[j] = __builtin_amdgcn_wmma_f32_16x16x32_bf16(
          false, a0, false, b, (short)0, acc[j], false, false);
      acc[4 + j] = __builtin_amdgcn_wmma_f32_16x16x32_bf16(
          false, a1, false, b, (short)0, acc[4 + j], false, false);
    }
    __syncthreads();  // all waves done reading buf before it is re-staged
    buf ^= 1;
  }

  // ---- epilogue: element i of v8f at lane L -> row = half*8+i, col = l16 ----
#pragma unroll
  for (int slab = 0; slab < 2; ++slab) {
#pragma unroll
    for (int j = 0; j < 4; ++j) {
      const v8f a = acc[slab * 4 + j];
      const int n = n0 + j * 16 + l16;
#pragma unroll
      for (int i = 0; i < 8; ++i) {
        const int m = m0 + wave * 32 + slab * 16 + half * 8 + i;
        float r = alpha * a[i];
        if (bias) r += bias[n];
        if (GELU) r = gelu_exact(r);
        if (rowscale) r *= rowscale[(long long)m * rsStride];
        float* cp = C + (long long)m * ldc + n;
        *cp = (beta == 0.0f) ? r : (beta * (*cp) + r);
      }
    }
  }
}

// ---------------------------------------------------------------------------
// Small fused kernels
// ---------------------------------------------------------------------------

__global__ void embed_gather(const int* __restrict__ src,
                             const float* __restrict__ embW,
                             float* __restrict__ emb, int T, int D) {
  long long i = (long long)blockIdx.x * blockDim.x + threadIdx.x;
  if (i >= (long long)T * D) return;
  const int t = (int)(i / D), d = (int)(i % D);
  emb[i] = embW[(long long)src[t] * D + d];
}

__global__ void mix_embed(const float* __restrict__ emb,
                          const float* __restrict__ t1,
                          const float* __restrict__ t2,
                          const float* __restrict__ qsP,
                          float* __restrict__ x, long long n) {
  long long i = (long long)blockIdx.x * blockDim.x + threadIdx.x;
  if (i >= n) return;
  const float qs = qsP[0];
  x[i] = emb[i] * (1.0f - qs) + t1[i] * qs + t2[i];
}

__global__ void residual_axpy(float* __restrict__ x, const float* __restrict__ a,
                              const float* __restrict__ b, long long n) {
  long long i = (long long)blockIdx.x * blockDim.x + threadIdx.x;
  if (i >= n) return;
  x[i] += a[i] + 0.1f * b[i];
}

__global__ __launch_bounds__(256) void softmax_rows(float* __restrict__ sc,
                                                    int rowlen) {
  float* p = sc + (long long)blockIdx.x * rowlen;
  const int tid = threadIdx.x;
  __shared__ float red[256];
  float mx = -3.4e38f;
  for (int i = tid; i < rowlen; i += 256) mx = fmaxf(mx, p[i]);
  red[tid] = mx;
  __syncthreads();
  for (int s = 128; s > 0; s >>= 1) {
    if (tid < s) red[tid] = fmaxf(red[tid], red[tid + s]);
    __syncthreads();
  }
  mx = red[0];
  __syncthreads();
  float sum = 0.0f;
  for (int i = tid; i < rowlen; i += 256) {
    const float e = __expf(p[i] - mx);
    p[i] = e;
    sum += e;
  }
  red[tid] = sum;
  __syncthreads();
  for (int s = 128; s > 0; s >>= 1) {
    if (tid < s) red[tid] += red[tid + s];
    __syncthreads();
  }
  const float inv = 1.0f / red[0];
  for (int i = tid; i < rowlen; i += 256) p[i] *= inv;
}

// gates = renorm( softmax(x@gw^T+gb) * (1 + 0.1*sigmoid(x@qw^T+qb)) ), E == 8
__global__ __launch_bounds__(256) void gates_kernel(
    const float* __restrict__ x, const float* __restrict__ gw,
    const float* __restrict__ gb, const float* __restrict__ qw,
    const float* __restrict__ qb, float* __restrict__ gates, int D) {
  const int E = 8;
  const int t = blockIdx.x;
  const int tid = threadIdx.x;
  const int g = tid >> 4;
  const int j = tid & 15;
  const float* xr = x + (long long)t * D;
  const float* wr = (g < E) ? (gw + (long long)g * D) : (qw + (long long)(g - E) * D);
  float s = 0.0f;
  for (int d = j; d < D; d += 16) s += xr[d] * wr[d];
  __shared__ float sm[16][16];
  __shared__ float dots[16];
  sm[g][j] = s;
  __syncthreads();
  if (j == 0) {
    float tot = 0.0f;
#pragma unroll
    for (int i = 0; i < 16; ++i) tot += sm[g][i];
    dots[g] = tot + ((g < E) ? gb[g] : qb[g - E]);
  }
  __syncthreads();
  if (tid == 0) {
    float mx = -3.4e38f;
    for (int e = 0; e < E; ++e) mx = fmaxf(mx, dots[e]);
    float ge[8], sum = 0.0f;
    for (int e = 0; e < E; ++e) { ge[e] = __expf(dots[e] - mx); sum += ge[e]; }
    float tot2 = 0.0f;
    for (int e = 0; e < E; ++e) {
      const float qg = 1.0f / (1.0f + __expf(-dots[E + e]));
      ge[e] = (ge[e] / sum) * (1.0f + 0.1f * qg);
      tot2 += ge[e];
    }
    for (int e = 0; e < E; ++e) gates[(long long)t * E + e] = ge[e] / tot2;
  }
}

__global__ __launch_bounds__(256) void add_ln(float* __restrict__ x,
                                              const float* __restrict__ moe,
                                              const float* __restrict__ g,
                                              const float* __restrict__ b,
                                              int D) {
  float* xr = x + (long long)blockIdx.x * D;
  const float* mr = moe + (long long)blockIdx.x * D;
  const int tid = threadIdx.x;
  __shared__ float r1[256], r2[256];
  float s = 0.0f, ss = 0.0f;
  for (int d = tid; d < D; d += 256) {
    const float v = xr[d] + mr[d];
    s += v;
    ss += v * v;
  }
  r1[tid] = s;
  r2[tid] = ss;
  __syncthreads();
  for (int k = 128; k > 0; k >>= 1) {
    if (tid < k) { r1[tid] += r1[tid + k]; r2[tid] += r2[tid + k]; }
    __syncthreads();
  }
  const float mean = r1[0] / (float)D;
  const float var = r2[0] / (float)D - mean * mean;
  const float inv = rsqrtf(var + 1e-5f);
  for (int d = tid; d < D; d += 256) {
    const float v = xr[d] + mr[d];
    xr[d] = (v - mean) * inv * g[d] + b[d];
  }
}

// ---------------------------------------------------------------------------
// Host-side orchestration
// ---------------------------------------------------------------------------

static void run_gemm(hipStream_t st, bool wt, bool gelu,
                     const float* A, const float* W, const float* bias,
                     const float* rs, int rsStride, float* C,
                     int M, int N, int K, int lda, int ldw, int ldc,
                     float alpha, float beta, int batch, int H2,
                     long long sAb, long long sAh, long long sWb, long long sWh,
                     long long sCb, long long sCh) {
  dim3 grid(N / BN, M / BM, batch), block(128);
  if (wt) {
    gemm_bf16_wmma<true, false><<<grid, block, 0, st>>>(
        A, W, bias, rs, rsStride, C, M, N, K, lda, ldw, ldc, alpha, beta, H2,
        sAb, sAh, sWb, sWh, sCb, sCh);
  } else if (gelu) {
    gemm_bf16_wmma<false, true><<<grid, block, 0, st>>>(
        A, W, bias, rs, rsStride, C, M, N, K, lda, ldw, ldc, alpha, beta, H2,
        sAb, sAh, sWb, sWh, sCb, sCh);
  } else {
    gemm_bf16_wmma<false, false><<<grid, block, 0, st>>>(
        A, W, bias, rs, rsStride, C, M, N, K, lda, ldw, ldc, alpha, beta, H2,
        sAb, sAh, sWb, sWh, sCb, sCh);
  }
}

extern "C" void kernel_launch(void* const* d_in, const int* in_sizes, int n_in,
                              void* d_out, int out_size, void* d_ws,
                              size_t ws_size, hipStream_t stream) {
  (void)in_sizes; (void)n_in; (void)out_size; (void)ws_size;
  const int B = 4, S = 512, D = 768, H = 12, L = 4, E = 8, F = 3072, V = 32000;
  const int HD = D / H, T = B * S, D3 = 3 * D;

  const int*   src     = (const int*)d_in[0];
  const float* qstate  = (const float*)d_in[1];
  const float* noise   = (const float*)d_in[2];
  const float* embW    = (const float*)d_in[3];
  const float* qprojW  = (const float*)d_in[4];
  const float* qprojB  = (const float*)d_in[5];
  const float* posW    = (const float*)d_in[6];
  const float* posB    = (const float*)d_in[7];
  const float* inW     = (const float*)d_in[8];
  const float* inB     = (const float*)d_in[9];
  const float* outW    = (const float*)d_in[10];
  const float* outB    = (const float*)d_in[11];
  const float* entW    = (const float*)d_in[12];
  const float* entB    = (const float*)d_in[13];
  const float* gateW   = (const float*)d_in[14];
  const float* gateB   = (const float*)d_in[15];
  const float* qgateW  = (const float*)d_in[16];
  const float* qgateB  = (const float*)d_in[17];
  const float* ew1     = (const float*)d_in[18];
  const float* eb1     = (const float*)d_in[19];
  const float* ew2     = (const float*)d_in[20];
  const float* eb2     = (const float*)d_in[21];
  const float* lnG     = (const float*)d_in[22];
  const float* lnB     = (const float*)d_in[23];
  const float* fcW     = (const float*)d_in[24];
  const float* fcB     = (const float*)d_in[25];

  float* ws   = (float*)d_ws;
  float* x    = ws;                                   // T*D
  float* tmp  = x + (long long)T * D;                 // T*D
  float* t1   = tmp + (long long)T * D;               // T*D
  float* t2   = t1 + (long long)T * D;                // T*D
  float* qkv  = t2 + (long long)T * D;                // T*3D
  float* sc   = qkv + (long long)T * D3;              // B*H*S*S
  float* o    = sc + (long long)B * H * S * S;        // T*D
  float* hbuf = o + (long long)T * D;                 // T*F
  float* moe  = hbuf + (long long)T * F;              // T*D
  float* gts  = moe + (long long)T * D;               // T*E

  const long long nTD = (long long)T * D;
  const int eltBlocks = (int)((nTD + 255) / 256);

  embed_gather<<<eltBlocks, 256, 0, stream>>>(src, embW, tmp, T, D);
  run_gemm(stream, false, false, tmp, qprojW, qprojB, nullptr, 0, t1,
           T, D, D, D, D, D, 1.0f, 0.0f, 1, 1, 0, 0, 0, 0, 0, 0);
  run_gemm(stream, false, false, noise, posW, posB, nullptr, 0, t2,
           T, D, D, D, D, D, 0.01f, 0.0f, 1, 1, 0, 0, 0, 0, 0, 0);
  mix_embed<<<eltBlocks, 256, 0, stream>>>(tmp, t1, t2, qstate, x, nTD);

  const float attScale = 1.0f / sqrtf((float)HD);

  for (int l = 0; l < L; ++l) {
    run_gemm(stream, false, false, x, inW + (long long)l * D3 * D,
             inB + (long long)l * D3, nullptr, 0, qkv,
             T, D3, D, D, D, D3, 1.0f, 0.0f, 1, 1, 0, 0, 0, 0, 0, 0);

    run_gemm(stream, false, false, qkv, qkv + D, nullptr, nullptr, 0, sc,
             S, S, HD, D3, D3, S, attScale, 0.0f, B * H, H,
             (long long)S * D3, HD, (long long)S * D3, HD,
             (long long)H * S * S, (long long)S * S);

    softmax_rows<<<B * H * S, 256, 0, stream>>>(sc, S);

    run_gemm(stream, true, false, sc, qkv + 2 * D, nullptr, nullptr, 0, o,
             S, HD, S, S, D3, D, 1.0f, 0.0f, B * H, H,
             (long long)H * S * S, (long long)S * S,
             (long long)S * D3, HD, (long long)S * D, HD);

    run_gemm(stream, false, false, o, outW + (long long)l * D * D,
             outB + (long long)l * D, nullptr, 0, tmp,
             T, D, D, D, D, D, 1.0f, 0.0f, 1, 1, 0, 0, 0, 0, 0, 0);
    run_gemm(stream, false, false, tmp, entW + (long long)l * D * D,
             entB + (long long)l * D, nullptr, 0, t1,
             T, D, D, D, D, D, 1.0f, 0.0f, 1, 1, 0, 0, 0, 0, 0, 0);
    residual_axpy<<<eltBlocks, 256, 0, stream>>>(x, tmp, t1, nTD);

    gates_kernel<<<T, 256, 0, stream>>>(
        x, gateW + (long long)l * E * D, gateB + (long long)l * E,
        qgateW + (long long)l * E * D, qgateB + (long long)l * E, gts, D);

    for (int e = 0; e < E; ++e) {
      const long long we = (long long)l * E + e;
      run_gemm(stream, false, true, x, ew1 + we * (long long)F * D,
               eb1 + we * F, nullptr, 0, hbuf,
               T, F, D, D, D, F, 1.0f, 0.0f, 1, 1, 0, 0, 0, 0, 0, 0);
      run_gemm(stream, false, false, hbuf, ew2 + we * (long long)D * F,
               eb2 + we * D, gts + e, E, moe,
               T, D, F, F, F, D, 1.0f, (e == 0) ? 0.0f : 1.0f,
               1, 1, 0, 0, 0, 0, 0, 0);
    }

    add_ln<<<T, 256, 0, stream>>>(x, moe, lnG + (long long)l * D,
                                  lnB + (long long)l * D, D);
  }

  run_gemm(stream, false, false, x, fcW, fcB, nullptr, 0, (float*)d_out,
           T, V, D, D, D, V, 1.0f, 0.0f, 1, 1, 0, 0, 0, 0, 0, 0);
}
